// CustomGRU_65420941853234
// MI455X (gfx1250) — matrix operand order repaired and checked by reading, not compiled, not run on hardware
//
#include <hip/hip_runtime.h>

typedef __attribute__((ext_vector_type(16))) _Float16 v16h;
typedef __attribute__((ext_vector_type(8)))  _Float16 v8h;
typedef __attribute__((ext_vector_type(8)))  float    v8f;

#define D    128
#define PS   136          // padded LDS row stride in halves (avoids bank conflicts)
#define NMAT 5            // Wz, Uz, (Wr+Ur), Wh^T, Uh^T

// ---------------------------------------------------------------------------
// Prep kernel 1: pack the 5 effective weight matrices into WMMA B-fragment
// layout, f16.  P[mat][nt(8)][kt(4)][lane(32)] = v16h, where element e of the
// fragment is Bmat[K][n] with K = 32*kt + 8*(lane>=16) + e + (e>=8 ? 8 : 0), n = 16*nt + lane%16.
// mat: 0=W_update, 1=U_update, 2=W_reset+U_reset, 3=W_h^T, 4=U_h^T
// ---------------------------------------------------------------------------
__global__ void pack_weights_kernel(const float* __restrict__ Wz,
                                    const float* __restrict__ Uz,
                                    const float* __restrict__ Wr,
                                    const float* __restrict__ Ur,
                                    const float* __restrict__ Wh,
                                    const float* __restrict__ Uh,
                                    _Float16* __restrict__ P) {
  int b    = blockIdx.x;        // 0..159 == (mat*8 + nt)*4 + kt
  int mat  = b >> 5;
  int lane = threadIdx.x;       // 0..31
  int kt   = b & 3;
  int nt   = (b >> 2) & 7;
  int n    = nt * 16 + (lane & 15);
  int kb   = kt * 32 + (lane >> 4) * 8;
  v16h out;
#pragma unroll
  for (int e = 0; e < 16; ++e) {
    int K = kb + e + (e >= 8 ? 8 : 0);
    float v;
    if      (mat == 0) v = Wz[K * D + n];
    else if (mat == 1) v = Uz[K * D + n];
    else if (mat == 2) v = Wr[K * D + n] + Ur[K * D + n];   // reset uses x twice
    else if (mat == 3) v = Wh[n * D + K];                    // x @ W_h^T
    else               v = Uh[n * D + K];                    // (r*h) @ U_h^T
    out[e] = (_Float16)v;
  }
  *((v16h*)P + ((size_t)b * 32 + lane)) = out;
}

// Prep kernel 2: bias vectors = column sums of the B matrices.
__global__ void bias_kernel(const float* __restrict__ Bz,
                            const float* __restrict__ Br,
                            const float* __restrict__ Bh,
                            float* __restrict__ bias) {
  int n = threadIdx.x;                                   // 0..127
  const float* src = (blockIdx.x == 0) ? Bz : (blockIdx.x == 1) ? Br : Bh;
  float s = 0.f;
  for (int i = 0; i < D; ++i) s += src[i * D + n];
  bias[blockIdx.x * D + n] = s;
}

// ---------------------------------------------------------------------------
// Main fused GRU kernel
// ---------------------------------------------------------------------------
// Hardware rcp (v_rcp_f32) instead of IEEE divide (avoids the
// div_scale/div_fmas/div_fixup expansion the compiler emits for '/').
__device__ __forceinline__ float fast_sigmoid(float t) {
  return __builtin_amdgcn_rcpf(1.0f + __expf(-t));
}
__device__ __forceinline__ float fast_tanh(float t) {
  return 2.0f * __builtin_amdgcn_rcpf(1.0f + __expf(-2.0f * t)) - 1.0f;
}

// Build one f16 A-fragment (16x32 layout) for this lane from an fp32 row.
// Lane needs floats K = 32*kt + 8*hl + {0..7}  and  K = 32*kt + 16 + 8*hl + {0..7}.
__device__ __forceinline__ v16h load_afrag(const float* __restrict__ row,
                                           int kt, int hl) {
  const float* p = row + kt * 32 + hl * 8;
  float4 a0 = *(const float4*)(p);
  float4 a1 = *(const float4*)(p + 4);
  float4 a2 = *(const float4*)(p + 16);
  float4 a3 = *(const float4*)(p + 20);
  v16h r;
  r[0]=(_Float16)a0.x;  r[1]=(_Float16)a0.y;  r[2]=(_Float16)a0.z;  r[3]=(_Float16)a0.w;
  r[4]=(_Float16)a1.x;  r[5]=(_Float16)a1.y;  r[6]=(_Float16)a1.z;  r[7]=(_Float16)a1.w;
  r[8]=(_Float16)a2.x;  r[9]=(_Float16)a2.y;  r[10]=(_Float16)a2.z; r[11]=(_Float16)a2.w;
  r[12]=(_Float16)a3.x; r[13]=(_Float16)a3.y; r[14]=(_Float16)a3.z; r[15]=(_Float16)a3.w;
  return r;
}

#define WMMA(a, b, c) \
  __builtin_amdgcn_wmma_f32_16x16x32_f16(false, (a), false, (b), (short)0, (c), false, false)

__global__ __launch_bounds__(256)
void gru_fused_kernel(const float* __restrict__ x, const float* __restrict__ h,
                      const _Float16* __restrict__ P, const float* __restrict__ bias,
                      float* __restrict__ out_ht, float* __restrict__ out_hc,
                      int ntiles) {
  __shared__ __align__(16) _Float16 lds_rh[2][16 * PS];

  const int tid  = threadIdx.x;
  const int w    = tid >> 5;          // wave id == output n-tile
  const int lane = tid & 31;
  const int li   = lane & 15;
  const int hl   = lane >> 4;
  const int ncol = w * 16 + li;       // this lane's output column (C-layout)

  const v16h* Pv = (const v16h*)P;

  // Register-cached B fragments for the two critical-path matrices only:
  // Wz (update gate, x-term) and Wr' = W_reset+U_reset (reset gate -> barrier).
  v16h Bz[4], Br[4];
#pragma unroll
  for (int kt = 0; kt < 4; ++kt) {
    Bz[kt] = Pv[((0 * 8 + w) * 4 + kt) * 32 + lane];
    Br[kt] = Pv[((2 * 8 + w) * 4 + kt) * 32 + lane];
  }
  const float bz = bias[ncol];
  const float br = bias[D + ncol];
  const float bh = bias[2 * D + ncol];

  for (int tile = blockIdx.x; tile < ntiles; tile += gridDim.x) {
    const int m0 = tile << 4;
    const float* xrow = x + (size_t)(m0 + li) * D;
    const float* hrow = h + (size_t)(m0 + li) * D;

    // Launder the packed-weight pointer so LICM cannot hoist the streamed
    // fragment loads out of the loop (they are L0-resident anyway, and
    // hoisting them would blow past 256 VGPRs).
    unsigned long long pbits = (unsigned long long)P;
    asm volatile("" : "+s"(pbits));
    const v16h* Ps = (const v16h*)pbits;

    // ---- A fragments of x; reset gate first (it feeds the barrier) -------
    v16h Ax[4];
#pragma unroll
    for (int kt = 0; kt < 4; ++kt) Ax[kt] = load_afrag(xrow, kt, hl);

    // r pre-activation: x@(Wr+Ur), two independent WMMA chains for ILP.
    v8f ar0 = {}, ar1 = {};
    ar0 = WMMA(Ax[0], Br[0], ar0);
    ar1 = WMMA(Ax[1], Br[1], ar1);
    ar0 = WMMA(Ax[2], Br[2], ar0);
    ar1 = WMMA(Ax[3], Br[3], ar1);

    // h in C-layout: rows m0+v+8*hl, column ncol (also reused in final blend).
    float hC[8];
#pragma unroll
    for (int v = 0; v < 8; ++v)
      hC[v] = h[(size_t)(m0 + v + 8 * hl) * D + ncol];

    // r = sigmoid(.); write r*h (f16) to LDS as early as possible so the
    // barrier can complete while we do the z / g3 matmuls below.
    _Float16* buf = lds_rh[tile & 1];
#pragma unroll
    for (int v = 0; v < 8; ++v) {
      float r = fast_sigmoid(ar0[v] + ar1[v] + br);
      buf[(v + 8 * hl) * PS + ncol] = (_Float16)(r * hC[v]);
    }

    // ---- Off-critical-path work before the barrier ------------------------
    // z pre-activation: x@Wz (cached) + h@Uz (streamed fragments).
    v8f accz = {};
#pragma unroll
    for (int kt = 0; kt < 4; ++kt) accz = WMMA(Ax[kt], Bz[kt], accz);

    v16h Ah[4];
#pragma unroll
    for (int kt = 0; kt < 4; ++kt) Ah[kt] = load_afrag(hrow, kt, hl);
#pragma unroll
    for (int kt = 0; kt < 4; ++kt) {
      v16h bu = Ps[((1 * 8 + w) * 4 + kt) * 32 + lane];
      accz = WMMA(Ah[kt], bu, accz);
    }

    // g3 = x @ Wh^T (streamed fragments).
    v8f accg3 = {};
#pragma unroll
    for (int kt = 0; kt < 4; ++kt) {
      v16h bw = Ps[((3 * 8 + w) * 4 + kt) * 32 + lane];
      accg3 = WMMA(Ax[kt], bw, accg3);
    }

    // Prefetch next tile's x/h rows while we sit at the barrier.
    int tnext = tile + gridDim.x;
    if (tnext < ntiles) {
      __builtin_prefetch(x + (size_t)((tnext << 4) + li) * D, 0, 3);
      __builtin_prefetch(h + (size_t)((tnext << 4) + li) * D, 0, 3);
    }

    __syncthreads();

    // g4 = (r*h) @ Uh^T : A fragments from LDS, Uh^T fragments streamed.
    v8f accg4 = {};
    const _Float16* rowp = buf + li * PS;
#pragma unroll
    for (int kt = 0; kt < 4; ++kt) {
      v8h l0 = *(const v8h*)(rowp + kt * 32 + 8 * hl);
      v8h l1 = *(const v8h*)(rowp + kt * 32 + 16 + 8 * hl);
      v16h arh = __builtin_shufflevector(l0, l1, 0, 1, 2, 3, 4, 5, 6, 7,
                                                 8, 9, 10, 11, 12, 13, 14, 15);
      v16h bu = Ps[((4 * 8 + w) * 4 + kt) * 32 + lane];
      accg4 = WMMA(arh, bu, accg4);
    }

    // Epilogue: z, h_cand, blend; stores in C-layout (2x64B segments / instr).
#pragma unroll
    for (int v = 0; v < 8; ++v) {
      float z  = fast_sigmoid(accz[v] + bz);
      float hc = fast_tanh(accg3[v] + accg4[v] + bh);
      float ht = hC[v] + z * (hc - hC[v]);          // (1-z)*h + z*hc
      size_t off = (size_t)(m0 + v + 8 * hl) * D + ncol;
      out_ht[off] = ht;
      out_hc[off] = hc;
    }
  }
}

// ---------------------------------------------------------------------------
extern "C" void kernel_launch(void* const* d_in, const int* in_sizes, int n_in,
                              void* d_out, int out_size, void* d_ws, size_t ws_size,
                              hipStream_t stream) {
  const float* x  = (const float*)d_in[0];
  const float* h  = (const float*)d_in[1];
  const float* Wz = (const float*)d_in[2];
  const float* Uz = (const float*)d_in[3];
  const float* Bz = (const float*)d_in[4];
  const float* Wr = (const float*)d_in[5];
  const float* Ur = (const float*)d_in[6];
  const float* Br = (const float*)d_in[7];
  const float* Wh = (const float*)d_in[8];
  const float* Uh = (const float*)d_in[9];
  const float* Bh = (const float*)d_in[10];

  const int Btot   = in_sizes[0] / D;     // 262144
  const int ntiles = Btot / 16;           // 16384

  _Float16* P    = (_Float16*)d_ws;
  float*    bias = (float*)((char*)d_ws + (size_t)NMAT * 8 * 4 * 32 * 16 * sizeof(_Float16));

  pack_weights_kernel<<<NMAT * 32, 32, 0, stream>>>(Wz, Uz, Wr, Ur, Wh, Uh, P);
  bias_kernel<<<3, 128, 0, stream>>>(Bz, Br, Bh, bias);

  float* out_ht = (float*)d_out;
  float* out_hc = out_ht + (size_t)Btot * D;

  int nblocks = ntiles < 2048 ? ntiles : 2048;
  gru_fused_kernel<<<nblocks, 256, 0, stream>>>(x, h, P, bias, out_ht, out_hc, ntiles);
}
